// RangeLSTM_85993835200590
// MI455X (gfx1250) — matrix operand order repaired and checked by reading, not compiled, not run on hardware
//
#include <hip/hip_runtime.h>
#include <hip/hip_bf16.h>

// LSTM: T=2048, B=32, I=H=512. Persistent 16-WG kernel, weights in LDS (bf16),
// per-step 32x2048x1024 GEMM via v_wmma_f32_16x16x32_bf16, device-wide sync per step.
// x is pre-converted to bf16 once per launch (if ws permits) and staged to LDS per
// step via async global->LDS copies.

#define T_STEPS 2048
#define BATCH   32
#define ISZ     512
#define HSZ     512
#define NWG     16     // 16 workgroups, each owns 32 hidden columns (128 gate rows)
#define NTHR    256    // 8 waves of 32

typedef __attribute__((ext_vector_type(16))) __bf16 v16bf;
typedef __attribute__((ext_vector_type(8)))  __bf16 v8bf;
typedef __attribute__((ext_vector_type(4)))  __bf16 v4bf;
typedef __attribute__((ext_vector_type(8)))  float  v8f;

union AFrag {
  v16bf  v;
  v8bf   g[2];
  __bf16 e[16];
};

__device__ __forceinline__ float sigmoidf_(float x) {
  return 1.0f / (1.0f + __expf(-x));
}

// One-shot f32 -> bf16 conversion of the whole x tensor (off the critical path).
__global__ void cvt_x_bf16(const float* __restrict__ x, __bf16* __restrict__ xbf, int n4) {
  const int i = blockIdx.x * blockDim.x + threadIdx.x;
  if (i < n4) {
    const float4 f = ((const float4*)x)[i];
    v4bf v;
    v[0] = (__bf16)f.x; v[1] = (__bf16)f.y;
    v[2] = (__bf16)f.z; v[3] = (__bf16)f.w;
    *(v4bf*)&xbf[i * 4] = v;
  }
}

__global__ __launch_bounds__(NTHR, 1)
void lstm_persistent(const float* __restrict__ x,
                     const float* __restrict__ w_ih,
                     const float* __restrict__ w_hh,
                     const float* __restrict__ bias,
                     float* __restrict__ out,
                     __bf16* __restrict__ hbuf,     // [BATCH][HSZ] bf16 in ws
                     unsigned* __restrict__ bar,    // grid barrier counter in ws
                     const __bf16* __restrict__ xbf) // bf16 x, or nullptr
{
  // 256 KB weight slice + 32 KB x_t stage + 16 KB gate tile = 304 KB (<= 320 KB/WGP)
  __shared__ __attribute__((aligned(32))) __bf16 Wlds[128 * 1024];
  __shared__ __attribute__((aligned(32))) __bf16 xls[BATCH * ISZ];
  __shared__ __attribute__((aligned(32))) float  gatesLds[BATCH * 128];

  const int tid  = threadIdx.x;
  const int wgid = blockIdx.x;                 // 0..15

  // ---- Prologue: load + permute + quantize this WG's weight slice into LDS ----
  // LDS row r (0..127): wave wv = r>>4 owns N-tile; within tile n' = jloc*4 + gate,
  // so each wave's 16 columns are its 4 hidden cols x {i,f,g,o}.
  for (int idx = tid; idx < 128 * 1024; idx += NTHR) {
    const int r    = idx >> 10;
    const int k    = idx & 1023;
    const int wv   = r >> 4;
    const int wi   = r & 15;
    const int jloc = wi >> 2;
    const int gate = wi & 3;                   // 0:i 1:f 2:g 3:o
    const int j    = wgid * 32 + wv * 4 + jloc;
    const int gr   = gate * HSZ + j;           // row of [w_ih; w_hh]
    const float w  = (k < ISZ) ? w_ih[(size_t)gr * ISZ + k]
                               : w_hh[(size_t)gr * HSZ + (k - ISZ)];
    Wlds[r * 1024 + k] = (__bf16)w;
  }
  // Zero this WG's slice of the global bf16 h buffer (h0 = 0).
  for (int idx = tid; idx < BATCH * 32; idx += NTHR) {
    const int bb = idx >> 5;
    const int jl = idx & 31;
    hbuf[bb * HSZ + wgid * 32 + jl] = (__bf16)0.0f;
  }

  // ---- Grid barrier (release/acquire, monotonic counter reset per launch) ----
  unsigned barCount = 0;
  auto grid_sync = [&]() {
    __syncthreads();
    if (tid == 0) {
      __threadfence();                              // release: flush WG's stores
      __builtin_amdgcn_s_cluster_barrier();         // NOP when not in a cluster
      __hip_atomic_fetch_add(bar, 1u, __ATOMIC_RELEASE, __HIP_MEMORY_SCOPE_AGENT);
      const unsigned target = (barCount + 1u) * (unsigned)NWG;
      while (__hip_atomic_load(bar, __ATOMIC_ACQUIRE, __HIP_MEMORY_SCOPE_AGENT) < target) {
        __builtin_amdgcn_s_sleep(2);
      }
      __threadfence();                              // acquire: invalidate WGP$
    }
    barCount++;
    __syncthreads();
  };
  grid_sync();

  // ---- Per-thread invariants ----
  const int lane = tid & 31;
  const int wv   = tid >> 5;                   // wave id 0..7 = N-tile
  const int lrow = lane & 15;
  const int hi   = lane >> 4;                  // lane half selector
  const int kOfs = hi * 8;
  const __bf16* wRow   = &Wlds[(wv * 16 + lrow) * 1024 + kOfs];
  const __bf16* xbase0 = &xls[(0 * 16 + lrow) * ISZ + kOfs];
  const __bf16* xbase1 = &xls[(1 * 16 + lrow) * ISZ + kOfs];
  const __bf16* hbase0 = hbuf + (0 * 16 + lrow) * HSZ + kOfs;
  const __bf16* hbase1 = hbuf + (1 * 16 + lrow) * HSZ + kOfs;

  // Elementwise role: thread owns one hidden col jl and 4 batches (bq0 + 8q).
  const int jl    = tid & 31;
  const int j     = wgid * 32 + jl;
  const int bq0   = tid >> 5;
  const int nIdx0 = (jl >> 2) * 16 + (jl & 3) * 4;
  const float bI = bias[0 * HSZ + j];
  const float bF = bias[1 * HSZ + j];
  const float bG = bias[2 * HSZ + j];
  const float bO = bias[3 * HSZ + j];
  float cReg[4] = {0.f, 0.f, 0.f, 0.f};

  // ---- Time loop ----
  for (int t = 0; t < T_STEPS; ++t) {
    // Stage x_t -> LDS.
    if (xbf) {
      // Async global->LDS 128-bit copies (ASYNCcnt-tracked); 128 B per thread.
      const __bf16* src = xbf + (size_t)t * BATCH * ISZ + tid * 64;
      const unsigned ldsOff = (unsigned)(uintptr_t)(&xls[tid * 64]);
      #pragma unroll
      for (int i = 0; i < 8; ++i) {
        asm volatile("global_load_async_to_lds_b128 %0, %1, off"
                     :: "v"(ldsOff + i * 16), "v"(src + i * 8) : "memory");
      }
      if (t + 1 < T_STEPS)
        __builtin_prefetch(xbf + (size_t)(t + 1) * BATCH * ISZ + tid * 128, 0, 0);
      asm volatile("s_wait_asynccnt 0x0" ::: "memory");
    } else {
      // Fallback: convert f32 -> bf16 on the fly.
      const float* xT = x + (size_t)t * BATCH * ISZ;
      for (int i4 = tid; i4 < (BATCH * ISZ) / 4; i4 += NTHR) {
        const float4 f = ((const float4*)xT)[i4];
        v4bf v;
        v[0] = (__bf16)f.x; v[1] = (__bf16)f.y;
        v[2] = (__bf16)f.z; v[3] = (__bf16)f.w;
        *(v4bf*)&xls[i4 * 4] = v;
      }
      if (t + 1 < T_STEPS)
        __builtin_prefetch(xT + BATCH * ISZ + tid * 64, 0, 0);
    }
    __syncthreads();

    // GEMM: gates[32 x 128-slice] = [x_t, h] @ Wslice^T, K = 1024, branch-free halves.
    v8f acc0 = {};
    v8f acc1 = {};
    #pragma unroll 4
    for (int kc = 0; kc < 16; ++kc) {          // x half: A from LDS
      AFrag bfrag, a0, a1;
      bfrag.g[0] = *(const v8bf*)(wRow + kc * 32);
      bfrag.g[1] = *(const v8bf*)(wRow + kc * 32 + 16);
      a0.g[0] = *(const v8bf*)(xbase0 + kc * 32);
      a0.g[1] = *(const v8bf*)(xbase0 + kc * 32 + 16);
      a1.g[0] = *(const v8bf*)(xbase1 + kc * 32);
      a1.g[1] = *(const v8bf*)(xbase1 + kc * 32 + 16);
      acc0 = __builtin_amdgcn_wmma_f32_16x16x32_bf16(false, a0.v, false, bfrag.v,
                                                     (short)0, acc0, false, false);
      acc1 = __builtin_amdgcn_wmma_f32_16x16x32_bf16(false, a1.v, false, bfrag.v,
                                                     (short)0, acc1, false, false);
    }
    #pragma unroll 4
    for (int kc = 0; kc < 16; ++kc) {          // h half: A from global bf16 h buffer
      AFrag bfrag, a0, a1;
      bfrag.g[0] = *(const v8bf*)(wRow + (16 + kc) * 32);
      bfrag.g[1] = *(const v8bf*)(wRow + (16 + kc) * 32 + 16);
      a0.g[0] = *(const v8bf*)(hbase0 + kc * 32);
      a0.g[1] = *(const v8bf*)(hbase0 + kc * 32 + 16);
      a1.g[0] = *(const v8bf*)(hbase1 + kc * 32);
      a1.g[1] = *(const v8bf*)(hbase1 + kc * 32 + 16);
      acc0 = __builtin_amdgcn_wmma_f32_16x16x32_bf16(false, a0.v, false, bfrag.v,
                                                     (short)0, acc0, false, false);
      acc1 = __builtin_amdgcn_wmma_f32_16x16x32_bf16(false, a1.v, false, bfrag.v,
                                                     (short)0, acc1, false, false);
    }

    // Spill D tiles to LDS (C/D layout: VGPR r -> M = r + 8*hi, N = lane&15).
    {
      const int n  = wv * 16 + lrow;
      const int mo = hi * 8;
      #pragma unroll
      for (int r = 0; r < 8; ++r) {
        gatesLds[(r + mo) * 128 + n]      = acc0[r];
        gatesLds[(16 + r + mo) * 128 + n] = acc1[r];
      }
    }
    __syncthreads();

    // Elementwise gate math; emit h (f32 -> out, bf16 -> hbuf), keep c in regs.
    float* outT = out + (size_t)t * BATCH * HSZ;
    #pragma unroll
    for (int q = 0; q < 4; ++q) {
      const int bb = bq0 + 8 * q;
      const float* gr4 = &gatesLds[bb * 128 + nIdx0];
      const float gi = sigmoidf_(gr4[0] + bI);
      const float gf = sigmoidf_(gr4[1] + bF);
      const float gg = tanhf(gr4[2] + bG);
      const float go = sigmoidf_(gr4[3] + bO);
      const float c  = gf * cReg[q] + gi * gg;
      cReg[q] = c;
      const float hv = go * tanhf(c);
      outT[bb * HSZ + j] = hv;
      hbuf[bb * HSZ + j] = (__bf16)hv;
      if (t == T_STEPS - 1) {
        const size_t base = (size_t)T_STEPS * BATCH * HSZ;
        out[base + bb * HSZ + j]               = hv;  // final h
        out[base + BATCH * HSZ + bb * HSZ + j] = c;   // final c
      }
    }
    grid_sync();   // publish h to all WGPs before next step
  }
}

extern "C" void kernel_launch(void* const* d_in, const int* in_sizes, int n_in,
                              void* d_out, int out_size, void* d_ws, size_t ws_size,
                              hipStream_t stream) {
  const float* x    = (const float*)d_in[0];   // [2048, 32, 512]
  const float* w_ih = (const float*)d_in[1];   // [2048, 512]
  const float* w_hh = (const float*)d_in[2];   // [2048, 512]
  const float* b    = (const float*)d_in[3];   // [2048]
  float* out = (float*)d_out;                  // outputs[T,B,H] ++ h[B,H] ++ c[B,H]

  unsigned* bar = (unsigned*)d_ws;                       // barrier counter
  __bf16*   hb  = (__bf16*)((char*)d_ws + 256);          // bf16 h buffer (32 KB)

  // Optional bf16 copy of x at ws+64KB (64 MB) if the workspace allows it.
  const size_t xElems  = (size_t)T_STEPS * BATCH * ISZ;
  const size_t xOff    = 65536;
  __bf16* xbf = nullptr;
  if (ws_size >= xOff + xElems * sizeof(__bf16)) {
    xbf = (__bf16*)((char*)d_ws + xOff);
    const int n4 = (int)(xElems / 4);
    cvt_x_bf16<<<(n4 + NTHR - 1) / NTHR, NTHR, 0, stream>>>(x, xbf, n4);
  }

  hipMemsetAsync(d_ws, 0, 256, stream);                  // reset barrier each launch
  lstm_persistent<<<NWG, NTHR, 0, stream>>>(x, w_ih, w_hh, b, out, hb, bar, xbf);
}
